// MultiHeadAttention_41747082117231
// MI455X (gfx1250) — matrix operand order repaired and checked by reading, not compiled, not run on hardware
//
#include <hip/hip_runtime.h>

// ---------------------------------------------------------------------------
// MultiHeadAttention forward for MI455X (gfx1250), wave32, bf16 WMMA.
// B=2, SQ=SK=1024, D=768, H=12, DH=64.
// ---------------------------------------------------------------------------

typedef __attribute__((ext_vector_type(16))) __bf16 v16bf;
typedef __attribute__((ext_vector_type(8)))  float  v8f;

#define DEV __device__ __forceinline__

constexpr int Bc  = 2;
constexpr int SQc = 1024;
constexpr int SKc = 1024;
constexpr int Dc  = 768;
constexpr int Hc  = 12;
// DH = 64

// ---- WMMA wrapper: D(f32 16x16) = A(bf16 16x32) x B(bf16 32x16) + C -------
DEV v8f wmma_bf16(v16bf a, v16bf b, v8f c) {
  return __builtin_amdgcn_wmma_f32_16x16x32_bf16(
      /*neg_a=*/false, a, /*neg_b=*/false, b,
      /*c_mod=*/(short)0, c, /*reuse_a=*/false, /*reuse_b=*/false);
}

// ---- Fragment loaders (ISA 7.12.2 layouts) --------------------------------
// A 16x32 bf16: lane l holds row l%16; g=l/16; elements e<8 -> k=8g+e,
// e>=8 -> k=16+8g+(e-8). Two contiguous 16B chunks per lane.
DEV v16bf load_a_frag(const __bf16* row, int g, int k0) {
  union { v16bf v; uint4 q[2]; } u;
  u.q[0] = *(const uint4*)(row + k0 + 8 * g);
  u.q[1] = *(const uint4*)(row + k0 + 16 + 8 * g);
  return u.v;
}
// B 32x16 bf16: lane l holds col l%16; elements e<8 -> k=8g+e,
// e>=8 -> k=16+8g+(e-8). Two contiguous 16B chunks per lane.
DEV v16bf load_b_frag(const __bf16* row, int g, int k0) {
  union { v16bf v; uint4 q[2]; } u;
  u.q[0] = *(const uint4*)(row + k0 + 8 * g);
  u.q[1] = *(const uint4*)(row + k0 + 16 + 8 * g);
  return u.v;
}

// ---- fp32 -> bf16 elementwise ---------------------------------------------
__global__ void cvt_f32_bf16(const float* __restrict__ in, void* outv, int n) {
  __bf16* out = (__bf16*)outv;
  int i = blockIdx.x * blockDim.x + threadIdx.x;
  int stride = gridDim.x * blockDim.x;
  for (; i < n; i += stride) out[i] = (__bf16)in[i];
}

// ---- fp32 W[K][N] -> bf16 Wt[N][K] (768x768) ------------------------------
__global__ void cvt_transpose_bf16(const float* __restrict__ W, void* outv) {
  __bf16* Wt = (__bf16*)outv;
  int i = blockIdx.x * blockDim.x + threadIdx.x;
  int stride = gridDim.x * blockDim.x;
  for (; i < Dc * Dc; i += stride) {
    int k = i / Dc, n = i % Dc;
    Wt[n * Dc + k] = (__bf16)W[i];
  }
}

// ---- WMMA GEMM: C[M=2048][N=768] = X[M][768] @ WtT + bias ------------------
// Wt is [N][K] row-major bf16.  Block = 64x64 tile, 4 waves (128 thr);
// wave w owns rows [16w,16w+16), iterates 4 n-subtiles. 96 WMMAs/wave.
// mode 0: out bf16 head-wise   [b][h][s][dh]
// mode 1: out bf16 head-wise-T [b][h][dh][s]   (for V)
// mode 2: out fp32 row-major   [m][n]          (final projection -> d_out)
__global__ __launch_bounds__(128) void gemm_bf16_k(
    const void* Xv_, const void* Wtv_, const float* __restrict__ bias,
    void* outv, int mode) {
  const __bf16* X  = (const __bf16*)Xv_;
  const __bf16* Wt = (const __bf16*)Wtv_;
  const int lane = threadIdx.x & 31;
  const int w    = threadIdx.x >> 5;
  const int g    = lane >> 4;
  const int cl   = lane & 15;
  const int mbase = blockIdx.y * 64 + w * 16;
  const int nb0   = blockIdx.x * 64;

  v8f acc[4] = {};
  const __bf16* arow = X + (size_t)(mbase + cl) * Dc;

  for (int kb = 0; kb < Dc; kb += 32) {
    v16bf af = load_a_frag(arow, g, kb);
#pragma unroll
    for (int t = 0; t < 4; ++t) {
      v16bf bf = load_b_frag(Wt + (size_t)(nb0 + 16 * t + cl) * Dc, g, kb);
      acc[t] = wmma_bf16(af, bf, acc[t]);
    }
  }

#pragma unroll
  for (int t = 0; t < 4; ++t) {
    const int n = nb0 + 16 * t + cl;
    const float bv = bias[n];
#pragma unroll
    for (int r = 0; r < 8; ++r) {
      const int m = mbase + 8 * g + r;        // C/D layout: row = 8g + r
      const float v = acc[t][r] + bv;
      if (mode == 2) {
        ((float*)outv)[(size_t)m * Dc + n] = v;
      } else {
        const int b = m >> 10, s = m & 1023;
        const int h = n >> 6,  dh = n & 63;
        __bf16* o = (__bf16*)outv;
        if (mode == 0)
          o[((size_t)(b * Hc + h) * SQc + s) * 64 + dh] = (__bf16)v;
        else
          o[((size_t)(b * Hc + h) * 64 + dh) * (size_t)SKc + s] = (__bf16)v;
      }
    }
  }
}

// ---- Flash attention: one wave per (b, h, 16 query rows) -------------------
// Q,K: bf16 [b][h][s][64]; Vt: bf16 [b][h][64][SK]; mask fp32 [B][H][SQ][SK];
// kpm int [B][SK] (False -> -inf).  ctx out: bf16 [b*SQ+q][768] row-major.
__global__ __launch_bounds__(32) void attn_k(
    const void* Qv, const void* Kv, const void* Vtv,
    const int* __restrict__ kpm, const float* __restrict__ mask, void* ctxv) {
  const __bf16* Q  = (const __bf16*)Qv;
  const __bf16* K  = (const __bf16*)Kv;
  const __bf16* Vt = (const __bf16*)Vtv;
  __bf16* ctx = (__bf16*)ctxv;

  const int lane = threadIdx.x;
  const int g = lane >> 4, cl = lane & 15;
  const int qbase = blockIdx.x * 16;
  const int h = blockIdx.y, b = blockIdx.z;

  const __bf16* Qh  = Q  + ((size_t)(b * Hc + h) * SQc + qbase) * 64;
  const __bf16* Kh  = K  + (size_t)(b * Hc + h) * SKc * 64;
  const __bf16* Vth = Vt + (size_t)(b * Hc + h) * 64 * SKc;
  const float*  mrow = mask + ((size_t)(b * Hc + h) * SQc + qbase) * (size_t)SKc;
  const int*    kp   = kpm + b * SKc;

  // Q fragment reused across all 64 K-tiles (2 k-steps of 32 over DH=64).
  const v16bf qf0 = load_a_frag(Qh + (size_t)cl * 64, g, 0);
  const v16bf qf1 = load_a_frag(Qh + (size_t)cl * 64, g, 32);

  v8f acc[4] = {};                // ctx accumulators, 4 dh-groups of 16
  float m8[8], l8[8];
#pragma unroll
  for (int r = 0; r < 8; ++r) { m8[r] = -1e30f; l8[r] = 0.f; }

  __shared__ __align__(16) __bf16 psh[16 * 32];  // P tile bounce buffer

  for (int kb = 0; kb < SKc; kb += 32) {
    // ---- S = Q K^T for two 16-col tiles -------------------------------
    v8f sa[2] = {};
#pragma unroll
    for (int t = 0; t < 2; ++t) {
      const __bf16* krow = Kh + (size_t)(kb + 16 * t + cl) * 64;
      sa[t] = wmma_bf16(qf0, load_b_frag(krow, g, 0),  sa[t]);
      sa[t] = wmma_bf16(qf1, load_b_frag(krow, g, 32), sa[t]);
    }

    const int pad0 = kp[kb + cl];
    const int pad1 = kp[kb + 16 + cl];

    float sv[2][8];
#pragma unroll
    for (int t = 0; t < 2; ++t) {
      const int pad = t ? pad1 : pad0;
#pragma unroll
      for (int r = 0; r < 8; ++r) {
        float s = sa[t][r] * 0.125f +                   // 1/sqrt(DH)
                  mrow[(size_t)(8 * g + r) * SKc + kb + 16 * t + cl];
        sv[t][r] = pad ? s : -1e30f;  // -1e30 avoids inf-inf NaN in rescale
      }
    }

    // ---- online softmax (rows live in 16-lane halves) -----------------
#pragma unroll
    for (int r = 0; r < 8; ++r) {
      float rm = fmaxf(sv[0][r], sv[1][r]);
      rm = fmaxf(rm, __shfl_xor(rm, 1, 16));
      rm = fmaxf(rm, __shfl_xor(rm, 2, 16));
      rm = fmaxf(rm, __shfl_xor(rm, 4, 16));
      rm = fmaxf(rm, __shfl_xor(rm, 8, 16));
      const float mn = fmaxf(m8[r], rm);
      const float sc = __expf(m8[r] - mn);
      m8[r] = mn;
      const float p0 = __expf(sv[0][r] - mn);
      const float p1 = __expf(sv[1][r] - mn);
      sv[0][r] = p0; sv[1][r] = p1;
      float rs = p0 + p1;
      rs += __shfl_xor(rs, 1, 16);
      rs += __shfl_xor(rs, 2, 16);
      rs += __shfl_xor(rs, 4, 16);
      rs += __shfl_xor(rs, 8, 16);
      l8[r] = l8[r] * sc + rs;
#pragma unroll
      for (int ng = 0; ng < 4; ++ng) acc[ng][r] *= sc;
    }

    // ---- reshape P: C/D layout -> A-fragment layout via LDS -----------
    __syncthreads();
#pragma unroll
    for (int t = 0; t < 2; ++t)
#pragma unroll
      for (int r = 0; r < 8; ++r)
        psh[(8 * g + r) * 32 + 16 * t + cl] = (__bf16)sv[t][r];
    __syncthreads();
    const v16bf pf = load_a_frag(psh + cl * 32, g, 0);

    // ---- ctx += P @ V (Vt rows are contiguous in kv) ------------------
#pragma unroll
    for (int ng = 0; ng < 4; ++ng)
      acc[ng] = wmma_bf16(pf, load_b_frag(Vth + (size_t)(16 * ng + cl) * SKc, g, kb),
                          acc[ng]);
  }

  // ---- normalize and store ctx as [b*SQ+q][h*64+dh] bf16 -----------------
#pragma unroll
  for (int ng = 0; ng < 4; ++ng)
#pragma unroll
    for (int r = 0; r < 8; ++r) {
      const float o = acc[ng][r] / l8[r];
      const int q = qbase + 8 * g + r;
      const int d = h * 64 + ng * 16 + cl;
      ctx[(size_t)(b * SQc + q) * Dc + d] = (__bf16)o;
    }
}

// ---------------------------------------------------------------------------
extern "C" void kernel_launch(void* const* d_in, const int* in_sizes, int n_in,
                              void* d_out, int out_size, void* d_ws, size_t ws_size,
                              hipStream_t stream) {
  const float* value = (const float*)d_in[0];
  const float* key   = (const float*)d_in[1];
  const float* query = (const float*)d_in[2];
  const int*   kpm   = (const int*)d_in[3];
  const float* amask = (const float*)d_in[4];
  const float* Wq = (const float*)d_in[5];
  const float* Wk = (const float*)d_in[6];
  const float* Wv = (const float*)d_in[7];
  const float* Wo = (const float*)d_in[8];
  const float* bq = (const float*)d_in[9];
  const float* bk = (const float*)d_in[10];
  const float* bv = (const float*)d_in[11];
  const float* bo = (const float*)d_in[12];

  char* ws = (char*)d_ws;
  const size_t E  = (size_t)2048 * 768 * 2;   // one [2048][768] bf16 buffer
  const size_t WB = (size_t)768 * 768 * 2;    // one 768x768 bf16 weight
  void* Xq  = ws + 0 * E;
  void* Xk  = ws + 1 * E;
  void* Xv  = ws + 2 * E;
  void* Wqt = ws + 3 * E + 0 * WB;
  void* Wkt = ws + 3 * E + 1 * WB;
  void* Wvt = ws + 3 * E + 2 * WB;
  void* Wot = ws + 3 * E + 3 * WB;
  void* Qh  = ws + 3 * E + 4 * WB + 0 * E;
  void* Kh  = ws + 3 * E + 4 * WB + 1 * E;
  void* Vth = ws + 3 * E + 4 * WB + 2 * E;
  void* Ctx = ws + 3 * E + 4 * WB + 3 * E;    // total ~26.8 MB

  const int n_x = 2048 * 768;
  cvt_f32_bf16<<<512, 256, 0, stream>>>(query, Xq, n_x);
  cvt_f32_bf16<<<512, 256, 0, stream>>>(key,   Xk, n_x);
  cvt_f32_bf16<<<512, 256, 0, stream>>>(value, Xv, n_x);
  cvt_transpose_bf16<<<576, 256, 0, stream>>>(Wq, Wqt);
  cvt_transpose_bf16<<<576, 256, 0, stream>>>(Wk, Wkt);
  cvt_transpose_bf16<<<576, 256, 0, stream>>>(Wv, Wvt);
  cvt_transpose_bf16<<<576, 256, 0, stream>>>(Wo, Wot);

  dim3 gg(Dc / 64, 2048 / 64);  // (12, 32)
  gemm_bf16_k<<<gg, 128, 0, stream>>>(Xq, Wqt, bq, Qh, 0);
  gemm_bf16_k<<<gg, 128, 0, stream>>>(Xk, Wkt, bk, Kh, 0);
  gemm_bf16_k<<<gg, 128, 0, stream>>>(Xv, Wvt, bv, Vth, 1);

  dim3 ga(SQc / 16, Hc, Bc);    // (64, 12, 2), one wave per block
  attn_k<<<ga, 32, 0, stream>>>(Qh, Kh, Vth, kpm, amask, Ctx);

  gemm_bf16_k<<<gg, 128, 0, stream>>>(Ctx, Wot, bo, d_out, 2);
}
